// PowerLawAttention_68934225100820
// MI455X (gfx1250) — matrix operand order repaired and checked by reading, not compiled, not run on hardware
//
#include <hip/hip_runtime.h>
#include <math.h>

typedef __attribute__((ext_vector_type(16))) _Float16 v16h;
typedef __attribute__((ext_vector_type(8)))  float    v8f;

#define B_     4
#define LQ_    1024
#define LK_    2048
#define D_     1024
#define H_     16
#define DK_    64
#define VOCAB_ 900

__device__ __forceinline__ void fill16(v16h& f, float4 a, float4 b, float4 c, float4 d)
{
    f[0]  = (_Float16)a.x; f[1]  = (_Float16)a.y; f[2]  = (_Float16)a.z; f[3]  = (_Float16)a.w;
    f[4]  = (_Float16)b.x; f[5]  = (_Float16)b.y; f[6]  = (_Float16)b.z; f[7]  = (_Float16)b.w;
    f[8]  = (_Float16)c.x; f[9]  = (_Float16)c.y; f[10] = (_Float16)c.z; f[11] = (_Float16)c.w;
    f[12] = (_Float16)d.x; f[13] = (_Float16)d.y; f[14] = (_Float16)d.z; f[15] = (_Float16)d.w;
}

__device__ __forceinline__ void fill16s(v16h& f, float4 a, float4 b, float4 c, float4 d, float s)
{
    f[0]  = (_Float16)(a.x * s); f[1]  = (_Float16)(a.y * s);
    f[2]  = (_Float16)(a.z * s); f[3]  = (_Float16)(a.w * s);
    f[4]  = (_Float16)(b.x * s); f[5]  = (_Float16)(b.y * s);
    f[6]  = (_Float16)(b.z * s); f[7]  = (_Float16)(b.w * s);
    f[8]  = (_Float16)(c.x * s); f[9]  = (_Float16)(c.y * s);
    f[10] = (_Float16)(c.z * s); f[11] = (_Float16)(c.w * s);
    f[12] = (_Float16)(d.x * s); f[13] = (_Float16)(d.y * s);
    f[14] = (_Float16)(d.z * s); f[15] = (_Float16)(d.w * s);
}

// ---------------------------------------------------------------------------
// Y[m,n] = sum_k X[m,k] * W[n,k] + bias[n]   (torch Linear: X @ W^T + b)
// Block: 256 threads (8 waves). Block tile 128(M) x 64(N), K stepped by 32,
// LDS double-buffered. Each wave owns 16(M) x 64(N) -> 4 WMMAs per K-step.
// ---------------------------------------------------------------------------
__global__ __launch_bounds__(256)
void proj_gemm_kernel(const float* __restrict__ X, const float* __restrict__ W,
                      const float* __restrict__ bias, float* __restrict__ Y,
                      int M, int N, int K)
{
    __shared__ _Float16 As[2][128 * 40];   // stride 40 halves (pad, 16B-friendly)
    __shared__ _Float16 Bs[2][64 * 40];

    const int tid  = threadIdx.x;
    const int lane = tid & 31;
    const int wave = tid >> 5;
    const int m0   = blockIdx.y * 128;
    const int n0   = blockIdx.x * 64;
    const int wm   = wave * 16;
    const int hi8  = (lane >= 16) ? 8 : 0;
    const int hi16 = (lane >= 16) ? 16 : 0;
    const int ml   = lane & 15;

    v8f acc[4] = {};

    // cooperative f32->f16 tile load; batch ALL global loads before converts
    auto load_tiles = [&](int k0, int buf) {
        float4 xa[4], wb[2];
#pragma unroll
        for (int i = 0; i < 4; ++i) {
            int t = tid + i * 256;
            int r = t >> 3, c4 = (t & 7) * 4;
            xa[i] = *(const float4*)&X[(size_t)(m0 + r) * K + (k0 + c4)];
        }
#pragma unroll
        for (int i = 0; i < 2; ++i) {
            int t = tid + i * 256;
            int r = t >> 3, c4 = (t & 7) * 4;
            wb[i] = *(const float4*)&W[(size_t)(n0 + r) * K + (k0 + c4)];
        }
#pragma unroll
        for (int i = 0; i < 4; ++i) {
            int t = tid + i * 256;
            int r = t >> 3, c4 = (t & 7) * 4;
            _Float16* p = &As[buf][r * 40 + c4];
            p[0] = (_Float16)xa[i].x; p[1] = (_Float16)xa[i].y;
            p[2] = (_Float16)xa[i].z; p[3] = (_Float16)xa[i].w;
        }
#pragma unroll
        for (int i = 0; i < 2; ++i) {
            int t = tid + i * 256;
            int r = t >> 3, c4 = (t & 7) * 4;
            _Float16* p = &Bs[buf][r * 40 + c4];
            p[0] = (_Float16)wb[i].x; p[1] = (_Float16)wb[i].y;
            p[2] = (_Float16)wb[i].z; p[3] = (_Float16)wb[i].w;
        }
    };

    load_tiles(0, 0);
    int cur = 0;
    for (int k0 = 0; k0 < K; k0 += 32) {
        __syncthreads();
        if (k0 + 32 < K) load_tiles(k0 + 32, cur ^ 1);  // overlap next tile

        // load ALL fragments first so LDS loads clause up, then 4 WMMAs
        v16h a, bf[4];
#pragma unroll
        for (int e = 0; e < 16; ++e) {
            int ka = hi16 + e;                         // A: contiguous k = hi16 + e (same order as B)
            a[e] = As[cur][(wm + ml) * 40 + ka];
        }
#pragma unroll
        for (int nt = 0; nt < 4; ++nt)
#pragma unroll
            for (int e = 0; e < 16; ++e)               // B: contiguous k = hi16 + e (same order as A)
                bf[nt][e] = Bs[cur][(nt * 16 + ml) * 40 + hi16 + e];
#pragma unroll
        for (int nt = 0; nt < 4; ++nt)
            acc[nt] = __builtin_amdgcn_wmma_f32_16x16x32_f16(
                false, a, false, bf[nt], (short)0, acc[nt], false, false);
        cur ^= 1;
    }

#pragma unroll
    for (int r = 0; r < 8; ++r) {                      // C: m = r + hi8, n = ml
        size_t row = (size_t)(m0 + wm + r + hi8) * N;
#pragma unroll
        for (int nt = 0; nt < 4; ++nt) {
            int n = n0 + nt * 16 + ml;
            Y[row + n] = acc[nt][r] + bias[n];
        }
    }
}

// ---------------------------------------------------------------------------
// Flash attention with learned bias + mask.
// Grid: B * (LQ/16) blocks. Block: 512 threads = 16 waves, wave w = head w.
// Per 32-key step: 4 WMMA for S = Q K^T, bias gather + mask + online softmax,
// P staged via per-wave LDS, 4 WMMA for O += P V.
// ---------------------------------------------------------------------------
__global__ __launch_bounds__(512)
void attn_kernel(const float* __restrict__ qh, const float* __restrict__ kh,
                 const float* __restrict__ vh, const int* __restrict__ b_idx,
                 const unsigned char* __restrict__ mask,
                 const float* __restrict__ b_table, float* __restrict__ out)
{
    __shared__ int           bidxLds[16 * 32];
    __shared__ unsigned char maskLds[16 * 32];
    __shared__ _Float16      pLds[16][16 * 40];

    const int tid  = threadIdx.x;
    const int lane = tid & 31;
    const int wave = tid >> 5;       // == head
    const int h    = wave;
    const int nqt  = LQ_ / 16;
    const int b    = blockIdx.x / nqt;
    const int q0   = (blockIdx.x % nqt) * 16;
    const int hi8  = (lane >= 16) ? 8 : 0;
    const int hi16 = (lane >= 16) ? 16 : 0;
    const int ml   = lane & 15;

    // Q fragments (A-layout), 1/sqrt(DK) folded in, float4 loads
    v16h qf0, qf1;
    {
        const float4* q4 = (const float4*)(qh + ((size_t)(b * LQ_ + q0 + ml)) * D_ + h * DK_);
        int o = hi16 >> 2;                             // 0 or 4 float4
        fill16s(qf0, q4[o], q4[o + 1], q4[o + 2], q4[o + 3], 0.125f);
        fill16s(qf1, q4[o + 8], q4[o + 9], q4[o + 10], q4[o + 11], 0.125f);
    }

    float mrow[8], lsum[8];
#pragma unroll
    for (int r = 0; r < 8; ++r) { mrow[r] = -3.0e38f; lsum[r] = 0.0f; }
    v8f o[4] = {};

    for (int kt = 0; kt < LK_ / 32; ++kt) {
        const int k0 = kt * 32;
        __syncthreads();                              // protect bidx/mask reuse
        {
            int r = tid >> 5, c = tid & 31;           // 512 elems, 512 threads
            size_t g = ((size_t)(b * LQ_ + q0 + r)) * LK_ + (k0 + c);
            bidxLds[tid] = b_idx[g];
            maskLds[tid] = mask[g];
        }
        __syncthreads();

        // K fragments (B-layout): two 16-key subtiles x two K=32 slices of DK
        v16h kf00, kf01, kf10, kf11;
        {
            const float4* k4a = (const float4*)(kh + ((size_t)(b * LK_ + k0 + ml)) * D_
                                                + h * DK_ + hi16);
            const float4* k4b = (const float4*)(kh + ((size_t)(b * LK_ + k0 + 16 + ml)) * D_
                                                + h * DK_ + hi16);
            fill16(kf00, k4a[0], k4a[1], k4a[2], k4a[3]);
            fill16(kf01, k4a[8], k4a[9], k4a[10], k4a[11]);
            fill16(kf10, k4b[0], k4b[1], k4b[2], k4b[3]);
            fill16(kf11, k4b[8], k4b[9], k4b[10], k4b[11]);
        }

        v8f sA = {}, sB = {};
        sA = __builtin_amdgcn_wmma_f32_16x16x32_f16(false, qf0, false, kf00, (short)0, sA, false, false);
        sA = __builtin_amdgcn_wmma_f32_16x16x32_f16(false, qf1, false, kf01, (short)0, sA, false, false);
        sB = __builtin_amdgcn_wmma_f32_16x16x32_f16(false, qf0, false, kf10, (short)0, sB, false, false);
        sB = __builtin_amdgcn_wmma_f32_16x16x32_f16(false, qf1, false, kf11, (short)0, sB, false, false);

        // bias + mask + online softmax (row = r + hi8, cols ml / 16+ml)
#pragma unroll
        for (int r = 0; r < 8; ++r) {
            int row = r + hi8;
            int ia = bidxLds[row * 32 + ml];
            int ib = bidxLds[row * 32 + 16 + ml];
            float va = sA[r] + b_table[ia * H_ + h];
            float vb = sB[r] + b_table[ib * H_ + h];
            if (!maskLds[row * 32 + ml])      va = -1.0e9f;
            if (!maskLds[row * 32 + 16 + ml]) vb = -1.0e9f;

            float mx = fmaxf(va, vb);
#pragma unroll
            for (int s = 1; s < 16; s <<= 1) mx = fmaxf(mx, __shfl_xor(mx, s, 32));
            float mnew  = fmaxf(mrow[r], mx);
            float scale = __expf(mrow[r] - mnew);
            float pa = __expf(va - mnew);
            float pb = __expf(vb - mnew);
            float ps = pa + pb;
#pragma unroll
            for (int s = 1; s < 16; s <<= 1) ps += __shfl_xor(ps, s, 32);
            lsum[r] = lsum[r] * scale + ps;
            mrow[r] = mnew;
#pragma unroll
            for (int d = 0; d < 4; ++d) o[d][r] *= scale;

            pLds[wave][row * 40 + ml]      = (_Float16)pa;
            pLds[wave][row * 40 + 16 + ml] = (_Float16)pb;
        }
        __syncthreads();   // order P store -> P load (block-uniform loop)

        // P as A-fragment (16x32 over keys), V as B-fragments (keys x dk);
        // load all V fragments first, then issue the 4 WMMAs back-to-back
        v16h pf, vf[4];
#pragma unroll
        for (int e = 0; e < 16; ++e) {
            int kc = hi16 + e;
            pf[e] = pLds[wave][ml * 40 + kc];
        }
        const float* vp = vh + ((size_t)(b * LK_ + k0 + hi16)) * D_ + h * DK_;
#pragma unroll
        for (int d = 0; d < 4; ++d)
#pragma unroll
            for (int e = 0; e < 16; ++e)
                vf[d][e] = (_Float16)vp[(size_t)e * D_ + d * 16 + ml];
#pragma unroll
        for (int d = 0; d < 4; ++d)
            o[d] = __builtin_amdgcn_wmma_f32_16x16x32_f16(false, pf, false, vf[d],
                                                          (short)0, o[d], false, false);
    }

    // finalize: divide by row sums, write [b, q, h*DK + dk]
    float* op = out + ((size_t)(b * LQ_ + q0)) * D_ + h * DK_;
#pragma unroll
    for (int r = 0; r < 8; ++r) {
        int row = r + hi8;
        float inv = 1.0f / lsum[r];
#pragma unroll
        for (int d = 0; d < 4; ++d)
            op[(size_t)row * D_ + d * 16 + ml] = o[d][r] * inv;
    }
}

// ---------------------------------------------------------------------------
extern "C" void kernel_launch(void* const* d_in, const int* in_sizes, int n_in,
                              void* d_out, int out_size, void* d_ws, size_t ws_size,
                              hipStream_t stream)
{
    const float*         q    = (const float*)d_in[0];
    const float*         k    = (const float*)d_in[1];
    const float*         v    = (const float*)d_in[2];
    const int*           bi   = (const int*)d_in[3];
    const unsigned char* mask = (const unsigned char*)d_in[4];
    const float*         Wq   = (const float*)d_in[5];
    const float*         bq   = (const float*)d_in[6];
    const float*         Wk   = (const float*)d_in[7];
    const float*         bk   = (const float*)d_in[8];
    const float*         Wv   = (const float*)d_in[9];
    const float*         bv   = (const float*)d_in[10];
    const float*         Wo   = (const float*)d_in[11];
    const float*         bo   = (const float*)d_in[12];
    const float*         bt   = (const float*)d_in[13];
    float*               out  = (float*)d_out;

    float* qh = (float*)d_ws;                      // [B, LQ, D]  16 MB
    float* kh = qh + (size_t)B_ * LQ_ * D_;        // [B, LK, D]  32 MB
    float* vh = kh + (size_t)B_ * LK_ * D_;        // [B, LK, D]  32 MB
    float* ao = vh + (size_t)B_ * LK_ * D_;        // [B, LQ, D]  16 MB

    dim3 blk(256);
    proj_gemm_kernel<<<dim3(D_ / 64, (B_ * LQ_) / 128), blk, 0, stream>>>(
        q, Wq, bq, qh, B_ * LQ_, D_, D_);
    proj_gemm_kernel<<<dim3(D_ / 64, (B_ * LK_) / 128), blk, 0, stream>>>(
        k, Wk, bk, kh, B_ * LK_, D_, D_);
    proj_gemm_kernel<<<dim3(D_ / 64, (B_ * LK_) / 128), blk, 0, stream>>>(
        v, Wv, bv, vh, B_ * LK_, D_, D_);

    attn_kernel<<<dim3(B_ * (LQ_ / 16)), dim3(512), 0, stream>>>(
        qh, kh, vh, bi, mask, bt, ao);

    proj_gemm_kernel<<<dim3(D_ / 64, (B_ * LQ_) / 128), blk, 0, stream>>>(
        ao, Wo, bo, out, B_ * LQ_, D_, D_);
}
